// MultiModalLoRA_proj_24206435680551
// MI455X (gfx1250) — compile-verified, matrix operand order checked by reading, not run on hardware
//
#include <hip/hip_runtime.h>
#include <hip/hip_bf16.h>

typedef _Float16 v16h __attribute__((ext_vector_type(16)));
typedef _Float16 v8h  __attribute__((ext_vector_type(8)));
typedef _Float16 v4h  __attribute__((ext_vector_type(4)));
typedef float    v8f  __attribute__((ext_vector_type(8)));

#define C_DIM 768           // channels (K and N)
#define RANK  16
#define M_MOD 4
#define TPM   8192          // tokens per modality (8 * 1024)
#define BM    128
#define BN    128
#define BK    32
#define KT    (C_DIM / BK)  // 24 k-steps
#define LDSS  40            // padded LDS row stride in halves (80B -> conflict-free)

// ---------------------------------------------------------------------------
// Kernel 1: fold shared + modality LoRA into the dense weight, convert to f16.
// W_eff[m][o][k] = Wp[o][k] + sum_r Bw[3][o][r]*A[3][r][k] + Bw[m][o][r]*A[m][r][k]
// ---------------------------------------------------------------------------
__global__ __launch_bounds__(256) void prep_weights_kernel(
    const float* __restrict__ Wp, const float* __restrict__ A,
    const float* __restrict__ Bw, _Float16* __restrict__ Weff)
{
  int idx = blockIdx.x * blockDim.x + threadIdx.x;
  if (idx >= M_MOD * C_DIM * C_DIM) return;
  int k = idx % C_DIM;
  int o = (idx / C_DIM) % C_DIM;
  int m = idx / (C_DIM * C_DIM);

  const float* As = A  + 3 * RANK * C_DIM;   // shared adapter (index 3)
  const float* Bs = Bw + 3 * C_DIM * RANK;
  const float* Am = A  + m * RANK * C_DIM;
  const float* Bm = Bw + m * C_DIM * RANK;

  float acc = Wp[o * C_DIM + k];
#pragma unroll
  for (int r = 0; r < RANK; ++r) {
    acc += Bs[o * RANK + r] * As[r * C_DIM + k];
    acc += Bm[o * RANK + r] * Am[r * C_DIM + k];
  }
  Weff[idx] = (_Float16)acc;
}

// ---------------------------------------------------------------------------
// Kernel 2: out[m] = x[m] (fp32->f16) @ W_eff[m]^T + bp, tiled WMMA GEMM.
// Block: 256 threads (8 waves). Block tile 128x128, K-step 32.
// Each wave computes a 64x32 sub-tile = 4x2 WMMA 16x16 accumulators.
// ---------------------------------------------------------------------------
__global__ __launch_bounds__(256) void lora_gemm_kernel(
    const float* __restrict__ x, const _Float16* __restrict__ Weff,
    const float* __restrict__ bp, float* __restrict__ out)
{
  __shared__ _Float16 Asub[2][BM][LDSS];  // 2*128*40*2B = 20 KB
  __shared__ _Float16 Bsub[2][BN][LDSS];  // 20 KB

  const int tid     = threadIdx.x;
  const int lane    = tid & 31;
  const int wave    = tid >> 5;       // 0..7
  const int waveM   = wave >> 2;      // 0..1 -> row offset waveM*64
  const int waveN   = wave & 3;       // 0..3 -> col offset waveN*32
  const int laneRow = lane & 15;      // row/col within 16x16 tile
  const int laneHi  = lane >> 4;      // 0/1 -> K sub-block {0..7,16..23} vs {8..15,24..31}

  const int  m        = blockIdx.z;
  const long rowBlk   = (long)blockIdx.y * BM;   // token tile within modality
  const int  colBlk   = blockIdx.x * BN;         // output-channel tile
  const long tokBase  = (long)m * TPM + rowBlk;

  const float*    xBase = x    + tokBase * C_DIM;
  const _Float16* wBase = Weff + (long)m * C_DIM * C_DIM + (long)colBlk * C_DIM;

  v8f acc[4][2] = {};

  float4 aReg[4];
  v4h    bReg[4];

  // ---- preload k-tile 0 into buffer 0 --------------------------------------
#pragma unroll
  for (int i = 0; i < 4; ++i) {
    int slot = tid + i * 256;         // 1024 slots: 128 rows x 8 chunks of 4
    int r  = slot >> 3;
    int c4 = slot & 7;
    aReg[i] = *(const float4*)(xBase + (long)r * C_DIM + c4 * 4);
    bReg[i] = *(const v4h*)  (wBase + (long)r * C_DIM + c4 * 4);
  }
#pragma unroll
  for (int i = 0; i < 4; ++i) {
    int slot = tid + i * 256;
    int r  = slot >> 3;
    int c4 = slot & 7;
    v4h ah;
    ah[0] = (_Float16)aReg[i].x; ah[1] = (_Float16)aReg[i].y;
    ah[2] = (_Float16)aReg[i].z; ah[3] = (_Float16)aReg[i].w;
    *(v4h*)(&Asub[0][r][c4 * 4]) = ah;
    *(v4h*)(&Bsub[0][r][c4 * 4]) = bReg[i];
  }
  __syncthreads();

  // ---- main K loop, double buffered ---------------------------------------
  for (int kt = 0; kt < KT; ++kt) {
    const int buf  = kt & 1;
    const bool more = (kt + 1) < KT;

    if (more) {
      const int kOff = (kt + 1) * BK;
#pragma unroll
      for (int i = 0; i < 4; ++i) {
        int slot = tid + i * 256;
        int r  = slot >> 3;
        int c4 = slot & 7;
        aReg[i] = *(const float4*)(xBase + (long)r * C_DIM + kOff + c4 * 4);
        bReg[i] = *(const v4h*)  (wBase + (long)r * C_DIM + kOff + c4 * 4);
      }
      if (kt + 3 < KT) {  // speculative prefetch 2 tiles further out
        int r = tid >> 3, c4 = tid & 7;
        __builtin_prefetch(xBase + (long)r * C_DIM + (kt + 3) * BK + c4 * 4, 0, 0);
      }
    }

    // ---- WMMA compute on current buffer -----------------------------------
    // Hoist ALL 12 ds_load_b128 fragment loads before the 8 WMMAs so the
    // loads can clause together and the WMMAs issue back-to-back (all 8
    // accumulators are independent -> no intra-group hazards).
    {
      v16h bfrag[2];
      v16h afrag[4];
#pragma unroll
      for (int ni = 0; ni < 2; ++ni) {
        const _Float16* p = &Bsub[buf][waveN * 32 + ni * 16 + laneRow][laneHi * 8];
        v8h lo = *(const v8h*)p;
        v8h hi = *(const v8h*)(p + 16);
        bfrag[ni] = __builtin_shufflevector(lo, hi,
            0, 1, 2, 3, 4, 5, 6, 7, 8, 9, 10, 11, 12, 13, 14, 15);
      }
#pragma unroll
      for (int fi = 0; fi < 4; ++fi) {
        const _Float16* p = &Asub[buf][waveM * 64 + fi * 16 + laneRow][laneHi * 8];
        v8h lo = *(const v8h*)p;
        v8h hi = *(const v8h*)(p + 16);
        afrag[fi] = __builtin_shufflevector(lo, hi,
            0, 1, 2, 3, 4, 5, 6, 7, 8, 9, 10, 11, 12, 13, 14, 15);
      }
#pragma unroll
      for (int fi = 0; fi < 4; ++fi) {
#pragma unroll
        for (int ni = 0; ni < 2; ++ni) {
          acc[fi][ni] = __builtin_amdgcn_wmma_f32_16x16x32_f16(
              /*neg_a=*/false, afrag[fi], /*neg_b=*/false, bfrag[ni],
              /*c_mod=*/(short)0, acc[fi][ni],
              /*reuse_a=*/false, /*reuse_b=*/false);
        }
      }
    }

    if (more) {
#pragma unroll
      for (int i = 0; i < 4; ++i) {
        int slot = tid + i * 256;
        int r  = slot >> 3;
        int c4 = slot & 7;
        v4h ah;
        ah[0] = (_Float16)aReg[i].x; ah[1] = (_Float16)aReg[i].y;
        ah[2] = (_Float16)aReg[i].z; ah[3] = (_Float16)aReg[i].w;
        *(v4h*)(&Asub[buf ^ 1][r][c4 * 4]) = ah;
        *(v4h*)(&Bsub[buf ^ 1][r][c4 * 4]) = bReg[i];
      }
    }
    __syncthreads();
  }

  // ---- epilogue: + bias, store fp32 ---------------------------------------
  // C/D layout: VGPR j -> M = j + 8*laneHi, N = laneRow.
#pragma unroll
  for (int ni = 0; ni < 2; ++ni) {
    const int col = colBlk + waveN * 32 + ni * 16 + laneRow;
    const float bias = bp[col];
#pragma unroll
    for (int fi = 0; fi < 4; ++fi) {
      const long row0 = tokBase + waveM * 64 + fi * 16 + laneHi * 8;
#pragma unroll
      for (int j = 0; j < 8; ++j) {
        out[(row0 + j) * C_DIM + col] = acc[fi][ni][j] + bias;
      }
    }
  }
}

// ---------------------------------------------------------------------------
extern "C" void kernel_launch(void* const* d_in, const int* in_sizes, int n_in,
                              void* d_out, int out_size, void* d_ws, size_t ws_size,
                              hipStream_t stream) {
  const float* x  = (const float*)d_in[0];   // [32, 1024, 768]
  const float* Wp = (const float*)d_in[1];   // [768, 768]
  const float* bp = (const float*)d_in[2];   // [768]
  const float* A  = (const float*)d_in[3];   // [4, 16, 768]
  const float* Bw = (const float*)d_in[4];   // [4, 768, 16]
  float* out = (float*)d_out;                // [32, 1024, 768]

  _Float16* Weff = (_Float16*)d_ws;          // 4*768*768*2 = 4.72 MB scratch

  // Fold LoRA into dense weights (f16).
  {
    int total  = M_MOD * C_DIM * C_DIM;
    int blocks = (total + 255) / 256;
    prep_weights_kernel<<<blocks, 256, 0, stream>>>(Wp, A, Bw, Weff);
  }

  // Batched WMMA GEMM with fused bias.
  {
    dim3 grid(C_DIM / BN, TPM / BM, M_MOD);  // (6, 64, 4)
    lora_gemm_kernel<<<grid, 256, 0, stream>>>(x, Weff, bp, out);
  }
}
